// DynamicAttentionMechanism_59768764891352
// MI455X (gfx1250) — matrix operand order repaired and checked by reading, not compiled, not run on hardware
//
#include <hip/hip_runtime.h>

// ---------------- problem constants ----------------
#define B_    4
#define CIN   64
#define CI    16
#define H_    127
#define W_    127
#define K_    7
#define S_    4
#define OH    31
#define OW    31
#define L_    961          // OH*OW
#define D_    784          // CI*49
#define TOPM  100
#define SCALE_ 10.0f

// ---------------- padded dims for WMMA tiling ----------------
#define MP    976          // L padded to 61*16  (A rows: p1 / attn)
#define NP1   1024         // p2 rows padded to 16*64 (N tiles of 64)
#define DP    800          // K pad GEMM1: 25*32
#define KP2   992          // K pad GEMM2: 31*32
#define NP3   832          // p3T rows padded to 13*64

typedef __attribute__((ext_vector_type(16))) __bf16 v16bf;
typedef __attribute__((ext_vector_type(8)))  float  v8f;

union Frag { v16bf v; uint4 q[2]; };

__device__ __forceinline__ unsigned short f2bf(float f) {
  unsigned u = __float_as_uint(f);
  unsigned r = u + 0x7FFFu + ((u >> 16) & 1u);   // round-to-nearest-even
  if ((u & 0x7F800000u) == 0x7F800000u) r = u;   // inf/nan passthrough
  return (unsigned short)(r >> 16);
}

// ---------------- front-end convs: b1 = conv3x3(x), b2/b3 = conv1x1(x) ----------------
__global__ void conv_front_kernel(const float* __restrict__ x,
    const float* __restrict__ gw, const float* __restrict__ gb,
    const float* __restrict__ tw, const float* __restrict__ tb,
    const float* __restrict__ pw, const float* __restrict__ pb,
    float* __restrict__ b1, float* __restrict__ b2, float* __restrict__ b3)
{
  int n = blockIdx.x * blockDim.x + threadIdx.x;
  const int HW = H_ * W_;
  if (n >= B_ * CI * HW) return;
  int w = n % W_;
  int h = (n / W_) % H_;
  int o = (n / HW) % CI;
  int b = n / (HW * CI);
  const float* xb = x + (size_t)b * CIN * HW;
  float s1 = gb[o], s2 = tb[o], s3 = pb[o];
  for (int c = 0; c < CIN; ++c) {
    const float* xc = xb + (size_t)c * HW;
    float xv = xc[h * W_ + w];
    s2 += tw[o * CIN + c] * xv;
    s3 += pw[o * CIN + c] * xv;
    const float* wg = gw + (size_t)(o * CIN + c) * 9;
    #pragma unroll
    for (int ki = 0; ki < 3; ++ki) {
      int hh = h - 1 + ki;
      if (hh < 0 || hh >= H_) continue;
      #pragma unroll
      for (int kj = 0; kj < 3; ++kj) {
        int ww = w - 1 + kj;
        if (ww < 0 || ww >= W_) continue;
        s1 += wg[ki * 3 + kj] * xc[hh * W_ + ww];
      }
    }
  }
  b1[n] = s1; b2[n] = s2; b3[n] = s3;
}

// ---------------- unfold + bf16 pack (zero-pads out-of-range) ----------------
// transposed==0: dst[b][r=l][c=d]   (rows x cols, row-major)
// transposed==1: dst[b][r=d][c=l]
__global__ void pack_kernel(const float* __restrict__ src, unsigned short* __restrict__ dst,
                            int rows, int cols, int transposed)
{
  int n = blockIdx.x * blockDim.x + threadIdx.x;
  int per = rows * cols;
  if (n >= B_ * per) return;
  int c = n % cols;
  int r = (n / cols) % rows;
  int b = n / per;
  int l = transposed ? c : r;
  int d = transposed ? r : c;
  float val = 0.f;
  if (l < L_ && d < D_) {
    int ch = d / 49, rem = d % 49;
    int ki = rem / 7, kj = rem % 7;
    int oh = l / OW, ow = l % OW;
    int h = oh * S_ + ki, w = ow * S_ + kj;
    val = src[(((size_t)b * CI + ch) * H_ + h) * W_ + w];
  }
  dst[(size_t)n] = f2bf(val);
}

// ---------------- batched bf16 WMMA GEMM: C[m,n] = sum_k A[m,k]*Bt[n,k] ----------------
// Wave tile: 16(M) x 64(N), K step 32, 4 accumulators, 8 waves/block.
__global__ void wmma_gemm_bf16_kernel(const unsigned short* __restrict__ A,
                                      const unsigned short* __restrict__ Bt,
                                      float* __restrict__ C,
                                      int K, int lda, int ldb, int ldc,
                                      int Mtiles, int Ntiles, int Mvalid, int Nvalid,
                                      unsigned long long aB, unsigned long long bB,
                                      unsigned long long cB)
{
  int wid  = blockIdx.x * (blockDim.x >> 5) + (threadIdx.x >> 5);
  int lane = threadIdx.x & 31;
  int per  = Mtiles * Ntiles;
  if (wid >= B_ * per) return;
  int b  = wid / per;
  int t  = wid % per;
  int m0 = (t / Ntiles) * 16;
  int n0 = (t % Ntiles) * 64;
  const unsigned short* Ab = A  + (size_t)b * aB;
  const unsigned short* Bb = Bt + (size_t)b * bB;
  float*                Cb = C  + (size_t)b * cB;
  int hl  = lane >> 4;     // lane half: selects contiguous K sub-range (k = 16*hl + s), same for A and B
  int l15 = lane & 15;
  const unsigned short* ar  = Ab + (size_t)(m0 + l15) * lda;
  const unsigned short* br0 = Bb + (size_t)(n0 +  0 + l15) * ldb;
  const unsigned short* br1 = Bb + (size_t)(n0 + 16 + l15) * ldb;
  const unsigned short* br2 = Bb + (size_t)(n0 + 32 + l15) * ldb;
  const unsigned short* br3 = Bb + (size_t)(n0 + 48 + l15) * ldb;

  v8f acc0 = {}, acc1 = {}, acc2 = {}, acc3 = {};
  for (int k0 = 0; k0 < K; k0 += 32) {
    Frag a, f0, f1, f2, f3;
    int ka = k0 + hl * 16;                // A: lane<16 -> K[0..16); lane>=16 -> K[16..32)
    a.q[0] = *(const uint4*)(ar + ka);
    a.q[1] = *(const uint4*)(ar + ka + 8);
    int kb = k0 + hl * 16;                // B: lane<16 -> K[0..16); lane>=16 -> K[16..32)
    f0.q[0] = *(const uint4*)(br0 + kb); f0.q[1] = *(const uint4*)(br0 + kb + 8);
    f1.q[0] = *(const uint4*)(br1 + kb); f1.q[1] = *(const uint4*)(br1 + kb + 8);
    f2.q[0] = *(const uint4*)(br2 + kb); f2.q[1] = *(const uint4*)(br2 + kb + 8);
    f3.q[0] = *(const uint4*)(br3 + kb); f3.q[1] = *(const uint4*)(br3 + kb + 8);
    acc0 = __builtin_amdgcn_wmma_f32_16x16x32_bf16(false, a.v, false, f0.v, (short)0, acc0, false, false);
    acc1 = __builtin_amdgcn_wmma_f32_16x16x32_bf16(false, a.v, false, f1.v, (short)0, acc1, false, false);
    acc2 = __builtin_amdgcn_wmma_f32_16x16x32_bf16(false, a.v, false, f2.v, (short)0, acc2, false, false);
    acc3 = __builtin_amdgcn_wmma_f32_16x16x32_bf16(false, a.v, false, f3.v, (short)0, acc3, false, false);
  }

  int r0 = m0 + hl * 8;                   // C layout: lane<16 rows m0+v, lane>=16 rows m0+8+v
  v8f accs[4] = {acc0, acc1, acc2, acc3};
  #pragma unroll
  for (int j = 0; j < 4; ++j) {
    int cidx = n0 + j * 16 + l15;
    if (cidx >= Nvalid) continue;
    #pragma unroll
    for (int v = 0; v < 8; ++v) {
      int r = r0 + v;
      if (r < Mvalid) Cb[(size_t)r * ldc + cidx] = accs[j][v];
    }
  }
}

// ---------------- per-row top-100 threshold + softmax -> bf16 attn ----------------
__global__ void topk_softmax_kernel(const float* __restrict__ score,
                                    unsigned short* __restrict__ attn)
{
  int row = blockIdx.x;                    // B_*MP rows
  int b = row / MP, l = row % MP;
  unsigned short* arow = attn + ((size_t)b * MP + l) * KP2;
  int tid = threadIdx.x;                   // 256 threads
  if (l >= L_) {                           // zero pad rows
    for (int i = tid; i < KP2; i += 256) arow[i] = 0;
    return;
  }
  const float* srow = score + ((size_t)b * MP + l) * MP;
  float v[4]; unsigned u[4];
  #pragma unroll
  for (int j = 0; j < 4; ++j) {
    int i = tid + j * 256;
    float f = 0.f; unsigned uu = 0u;
    if (i < L_) {
      f = srow[i];
      unsigned xb = __float_as_uint(f);
      uu = (xb & 0x80000000u) ? ~xb : (xb | 0x80000000u);  // monotonic key
    }
    v[j] = f; u[j] = uu;
  }
  __shared__ int scnt;
  __shared__ float red[256];
  // binary search for largest T with count(u >= T) >= TOPM
  unsigned lo = 0u, hi = 0xFFFFFFFFu;
  while (lo < hi) {
    unsigned mid = (unsigned)((((unsigned long long)lo + hi) + 1ull) >> 1);
    if (tid == 0) scnt = 0;
    __syncthreads();
    int c = 0;
    #pragma unroll
    for (int j = 0; j < 4; ++j) c += (u[j] >= mid) ? 1 : 0;
    if (c) atomicAdd(&scnt, c);
    __syncthreads();
    int cnt = scnt;
    __syncthreads();
    if (cnt >= TOPM) lo = mid; else hi = mid - 1;
  }
  unsigned T = lo;
  // sparse row = top values, zeros elsewhere; softmax(sparse * SCALE)
  float sv[4]; float lmax = 0.f;
  #pragma unroll
  for (int j = 0; j < 4; ++j) {
    int i = tid + j * 256;
    float s = (i < L_ && u[j] >= T) ? v[j] : 0.f;
    sv[j] = s;
    if (i < L_) lmax = fmaxf(lmax, s);     // zeros included -> max >= 0, matches ref
  }
  red[tid] = lmax; __syncthreads();
  for (int st = 128; st > 0; st >>= 1) { if (tid < st) red[tid] = fmaxf(red[tid], red[tid + st]); __syncthreads(); }
  float m = red[0];
  __syncthreads();
  float e[4]; float lsum = 0.f;
  #pragma unroll
  for (int j = 0; j < 4; ++j) {
    int i = tid + j * 256;
    float ee = 0.f;
    if (i < L_) { ee = __expf((sv[j] - m) * SCALE_); lsum += ee; }
    e[j] = ee;
  }
  red[tid] = lsum; __syncthreads();
  for (int st = 128; st > 0; st >>= 1) { if (tid < st) red[tid] += red[tid + st]; __syncthreads(); }
  float inv = 1.f / red[0];
  #pragma unroll
  for (int j = 0; j < 4; ++j) {
    int i = tid + j * 256;
    if (i < L_) arow[i] = f2bf(e[j] * inv);
  }
  for (int i = L_ + tid; i < KP2; i += 256) arow[i] = 0;   // zero pad cols
}

// ---------------- fold (overlap-add) + mask divide + restore 1x1 conv ----------------
__global__ void fold_restore_kernel(const float* __restrict__ agg,
    const float* __restrict__ mw, const float* __restrict__ mb,
    const float* __restrict__ rw, const float* __restrict__ rb,
    float* __restrict__ out)
{
  __shared__ float S[CI], MB[CI];
  int tid = threadIdx.x;
  if (tid < CI) {
    float s = 0.f;
    for (int c = 0; c < CIN; ++c) s += mw[tid * CIN + c];
    S[tid] = s; MB[tid] = mb[tid];
  }
  __syncthreads();
  int n = blockIdx.x * blockDim.x + tid;
  if (n >= B_ * H_ * W_) return;
  int w = n % W_, h = (n / W_) % H_, b = n / (H_ * W_);
  float fold[CI];
  #pragma unroll
  for (int c = 0; c < CI; ++c) fold[c] = 0.f;
  // windows covering (h,w): oh in {h/4-1, h/4} with 0<=h-4*oh<=6 (stride 4, k 7)
  int ohv[2], kiv[2], rh = 0;
  int oh0 = h >> 2;
  for (int dh = -1; dh <= 0; ++dh) {
    int oh = oh0 + dh;
    if (oh < 0 || oh >= OH) continue;
    int ki = h - oh * S_;
    if (ki < 0 || ki > K_ - 1) continue;
    ohv[rh] = oh; kiv[rh] = ki; ++rh;
  }
  int owv[2], kjv[2], rwc = 0;
  int ow0 = w >> 2;
  for (int dw = -1; dw <= 0; ++dw) {
    int ow = ow0 + dw;
    if (ow < 0 || ow >= OW) continue;
    int kj = w - ow * S_;
    if (kj < 0 || kj > K_ - 1) continue;
    owv[rwc] = ow; kjv[rwc] = kj; ++rwc;
  }
  for (int a = 0; a < rh; ++a)
    for (int q = 0; q < rwc; ++q) {
      int l = ohv[a] * OW + owv[q];
      int dbase = kiv[a] * K_ + kjv[q];
      const float* ar = agg + ((size_t)b * L_ + l) * D_;
      #pragma unroll
      for (int c = 0; c < CI; ++c) fold[c] += ar[c * 49 + dbase];
    }
  float cnt = (float)(rh * rwc);           // fold(unfold(ones)) coverage count
  float o16[CI];
  #pragma unroll
  for (int c = 0; c < CI; ++c) o16[c] = fold[c] / (cnt * S[c] + MB[c] + 1e-8f);
  size_t obase = (size_t)b * CIN * H_ * W_ + (size_t)h * W_ + w;
  for (int oc = 0; oc < CIN; ++oc) {
    float a2 = rb[oc];
    #pragma unroll
    for (int c = 0; c < CI; ++c) a2 += rw[oc * CI + c] * o16[c];
    out[obase + (size_t)oc * H_ * W_] = a2;
  }
}

// ---------------- host-side launch ----------------
extern "C" void kernel_launch(void* const* d_in, const int* in_sizes, int n_in,
                              void* d_out, int out_size, void* d_ws, size_t ws_size,
                              hipStream_t stream)
{
  (void)in_sizes; (void)n_in; (void)out_size; (void)ws_size;
  const float* x  = (const float*)d_in[0];
  const float* gw = (const float*)d_in[1];
  const float* gb = (const float*)d_in[2];
  const float* tw = (const float*)d_in[3];
  const float* tb = (const float*)d_in[4];
  const float* pw = (const float*)d_in[5];
  const float* pb = (const float*)d_in[6];
  const float* mw = (const float*)d_in[7];
  const float* mb = (const float*)d_in[8];
  const float* rw = (const float*)d_in[9];
  const float* rb = (const float*)d_in[10];
  float* out = (float*)d_out;

  char* ws = (char*)d_ws;
  size_t off = 0;
  auto take = [&](size_t bytes) -> char* {
    char* p = ws + off;
    off += (bytes + 255) & ~(size_t)255;
    return p;
  };
  float*          b1    = (float*)take((size_t)B_ * CI * H_ * W_ * 4);
  float*          b2    = (float*)take((size_t)B_ * CI * H_ * W_ * 4);
  float*          b3    = (float*)take((size_t)B_ * CI * H_ * W_ * 4);
  unsigned short* p1    = (unsigned short*)take((size_t)B_ * MP  * DP  * 2);
  unsigned short* p2    = (unsigned short*)take((size_t)B_ * NP1 * DP  * 2);
  unsigned short* p3t   = (unsigned short*)take((size_t)B_ * NP3 * KP2 * 2);
  float*          score = (float*)take((size_t)B_ * MP * MP * 4);
  unsigned short* attn  = (unsigned short*)take((size_t)B_ * MP * KP2 * 2);
  float*          agg   = (float*)take((size_t)B_ * L_ * D_ * 4);

  // 1) front-end convs
  int nconv = B_ * CI * H_ * W_;
  conv_front_kernel<<<(nconv + 255) / 256, 256, 0, stream>>>(x, gw, gb, tw, tb, pw, pb, b1, b2, b3);

  // 2) unfold + bf16 pack (also writes the zero padding)
  pack_kernel<<<((B_ * MP  * DP ) + 255) / 256, 256, 0, stream>>>(b1, p1,  MP,  DP,  0);
  pack_kernel<<<((B_ * NP1 * DP ) + 255) / 256, 256, 0, stream>>>(b2, p2,  NP1, DP,  0);
  pack_kernel<<<((B_ * NP3 * KP2) + 255) / 256, 256, 0, stream>>>(b3, p3t, NP3, KP2, 1);

  // 3) GEMM1: score[b] = p1 (961x784) * p2^T  -> [961x961], K padded to 800
  int w1 = B_ * 61 * 16;   // waves (16x64 tiles)
  wmma_gemm_bf16_kernel<<<(w1 + 7) / 8, 256, 0, stream>>>(
      p1, p2, score, DP, DP, DP, MP, 61, 16, L_, L_,
      (unsigned long long)MP * DP, (unsigned long long)NP1 * DP, (unsigned long long)MP * MP);

  // 4) top-100 threshold + softmax -> bf16 attn [MP x KP2] (zero padded)
  topk_softmax_kernel<<<B_ * MP, 256, 0, stream>>>(score, attn);

  // 5) GEMM2: agg[b] = attn (961x961) * p3 (961x784), K padded to 992
  int w2 = B_ * 61 * 13;
  wmma_gemm_bf16_kernel<<<(w2 + 7) / 8, 256, 0, stream>>>(
      attn, p3t, agg, KP2, KP2, KP2, D_, 61, 13, L_, D_,
      (unsigned long long)MP * KP2, (unsigned long long)NP3 * KP2, (unsigned long long)L_ * D_);

  // 6) fold + mask divide + restore conv -> output
  int nf = B_ * H_ * W_;
  fold_restore_kernel<<<(nf + 255) / 256, 256, 0, stream>>>(agg, mw, mb, rw, rb, out);
}
